// CovarianceSimilarity_50319836839983
// MI455X (gfx1250) — matrix operand order repaired and checked by reading, not compiled, not run on hardware
//
#include <hip/hip_runtime.h>

// ---------------------------------------------------------------------------
// CovarianceSimilarity for MI455X (gfx1250, wave32, WMMA).
// Compute-bound (~1200 FLOP/byte): split-precision bf16 WMMA (hi/lo, 3
// products) for near-fp32 accuracy.  Operands are stored pre-swizzled in the
// contiguous-k operand order used here (planar hi/lo, row-major cov rows, transposed
// Qn planes in LDS) so the hot loop loads whole v16bf vectors with b128 ops
// and does zero per-element unpacking.
// ---------------------------------------------------------------------------

constexpr int NQ   = 75;
constexpr int C    = 640;
constexpr int Dd   = 1024;          // H*W
constexpr int WAY  = 5;
constexpr int SHOT = 5;
constexpr int NS   = SHOT * Dd;     // 5120 samples per covariance row
constexpr float NEG_SLOPE = 0.2f;
constexpr float EPSF      = 1e-8f;

typedef __bf16 bf16_t;
typedef __attribute__((ext_vector_type(16))) __bf16 v16bf;
typedef __attribute__((ext_vector_type(8)))  float  v8f;

// bf16 round-to-nearest-even via bit ops
__device__ __forceinline__ unsigned short f32_to_bf16_rne(float f) {
  unsigned u = __builtin_bit_cast(unsigned, f);
  u += 0x7FFFu + ((u >> 16) & 1u);
  return (unsigned short)(u >> 16);
}
__device__ __forceinline__ float bf16_bits_to_f32(unsigned short h) {
  return __builtin_bit_cast(float, ((unsigned)h) << 16);
}
__device__ __forceinline__ bf16_t bf16_from_bits(unsigned short h) {
  return __builtin_bit_cast(bf16_t, h);
}

// ---------------------------------------------------------------------------
// K0a: per-(w,c) mean over Shot*d samples.  grid = WAY*C blocks of 256.
// ---------------------------------------------------------------------------
__global__ __launch_bounds__(256) void mean_kernel(const float* __restrict__ support,
                                                   float* __restrict__ mean) {
  __shared__ float red[256];
  const int wc = blockIdx.x;
  const int w = wc / C, c = wc - w * C;
  float acc = 0.0f;
  for (int s = 0; s < SHOT; ++s) {
    const float* p = support + ((size_t)(w * SHOT + s) * C + c) * Dd;
    for (int i = threadIdx.x; i < Dd; i += 256) acc += p[i];
  }
  red[threadIdx.x] = acc;
  __syncthreads();
  for (int s = 128; s > 0; s >>= 1) {
    if (threadIdx.x < s) red[threadIdx.x] += red[threadIdx.x + s];
    __syncthreads();
  }
  if (threadIdx.x == 0) mean[wc] = red[0] * (1.0f / (float)NS);
}

// ---------------------------------------------------------------------------
// K0b: per-(n,c) 1/(||row||+eps).  grid = NQ*C blocks of 256.
// ---------------------------------------------------------------------------
__global__ __launch_bounds__(256) void norm_kernel(const float* __restrict__ query,
                                                   float* __restrict__ inv_norm) {
  __shared__ float red[256];
  const int nc = blockIdx.x;
  const float* p = query + (size_t)nc * Dd;
  float acc = 0.0f;
  for (int i = threadIdx.x; i < Dd; i += 256) { float x = p[i]; acc += x * x; }
  red[threadIdx.x] = acc;
  __syncthreads();
  for (int s = 128; s > 0; s >>= 1) {
    if (threadIdx.x < s) red[threadIdx.x] += red[threadIdx.x + s];
    __syncthreads();
  }
  if (threadIdx.x == 0) inv_norm[nc] = 1.0f / (sqrtf(red[0]) + EPSF);
}

// ---------------------------------------------------------------------------
// K1: cov[w] = S_c S_c^T / (N-1+eps) via split-bf16 WMMA.
// One wave per 16x16 output tile; K = 5120 in chunks of 32.
// Output written as planar hi/lo bf16 with columns in natural (unpermuted) order.
// K2 then loads A operands as contiguous
// 32-byte vectors.
// grid = WAY*40*40/8 = 1000 blocks of 256 (8 waves).
// ---------------------------------------------------------------------------
__global__ __launch_bounds__(256) void cov_kernel(const float* __restrict__ support,
                                                  const float* __restrict__ mean,
                                                  bf16_t* __restrict__ cov_hi,
                                                  bf16_t* __restrict__ cov_lo) {
  const int tid  = threadIdx.x;
  const int wave = tid >> 5;
  const int lane = tid & 31;
  const int half = lane >> 4;
  const int l16  = lane & 15;

  const int gidx = blockIdx.x * 8 + wave;          // 0..7999
  const int w    = gidx / 1600;
  const int rem  = gidx - w * 1600;
  const int ti   = rem / 40;                       // M tile (row c)
  const int tj   = rem - ti * 40;                  // N tile (col b)

  const int cA = ti * 16 + l16;                    // A row (M = l16)
  const int cB = tj * 16 + l16;                    // B col (N = l16)
  const float meanA = mean[w * C + cA];
  const float meanB = mean[w * C + cB];
  const float* supW = support + (size_t)w * SHOT * C * Dd;

  v8f acc = {};
  for (int kb = 0; kb < NS / 32; ++kb) {
    const int kbase = kb * 32;
    v16bf a_hi, a_lo, b_hi, b_lo;
    // A 16x32 bf16 layout: elem e -> K = 16*half + e
    {
      const int n0 = kbase + half * 16;
      const int n1 = n0 + 8;
      const float* pA0 = supW + ((size_t)((n0 >> 10) * C + cA) << 10) + (n0 & 1023);
      const float* pA1 = supW + ((size_t)((n1 >> 10) * C + cA) << 10) + (n1 & 1023);
#pragma unroll
      for (int e = 0; e < 8; ++e) {
        float x0 = pA0[e] - meanA;
        float x1 = pA1[e] - meanA;
        unsigned short h0 = f32_to_bf16_rne(x0);
        unsigned short l0 = f32_to_bf16_rne(x0 - bf16_bits_to_f32(h0));
        unsigned short h1 = f32_to_bf16_rne(x1);
        unsigned short l1 = f32_to_bf16_rne(x1 - bf16_bits_to_f32(h1));
        a_hi[e]     = bf16_from_bits(h0);  a_lo[e]     = bf16_from_bits(l0);
        a_hi[e + 8] = bf16_from_bits(h1);  a_lo[e + 8] = bf16_from_bits(l1);
      }
    }
    // B 32x16 bf16 layout: elem e -> K = 16*half + e
    {
      const int nb = kbase + half * 16;
      const float* pB = supW + ((size_t)((nb >> 10) * C + cB) << 10) + (nb & 1023);
#pragma unroll
      for (int e = 0; e < 16; ++e) {
        float x = pB[e] - meanB;
        unsigned short h = f32_to_bf16_rne(x);
        unsigned short l = f32_to_bf16_rne(x - bf16_bits_to_f32(h));
        b_hi[e] = bf16_from_bits(h);  b_lo[e] = bf16_from_bits(l);
      }
    }
    acc = __builtin_amdgcn_wmma_f32_16x16x32_bf16(false, a_hi, false, b_hi, (short)0, acc, false, false);
    acc = __builtin_amdgcn_wmma_f32_16x16x32_bf16(false, a_hi, false, b_lo, (short)0, acc, false, false);
    acc = __builtin_amdgcn_wmma_f32_16x16x32_bf16(false, a_lo, false, b_hi, (short)0, acc, false, false);
  }

  const float scale = 1.0f / ((float)(NS - 1) + EPSF);
  // Column position for cB: unpermuted (o, ph, prun unused)
  const int o    = cB & 31;
  const int ph   = (o >> 3) & 1;
  const int prun = o >> 4;
  const int pcol = cB;
  bf16_t* dhi = cov_hi + (size_t)w * C * C;
  bf16_t* dlo = cov_lo + (size_t)w * C * C;
#pragma unroll
  for (int r = 0; r < 8; ++r) {                    // C/D: elem r -> M = r + 8*half
    const int crow = ti * 16 + half * 8 + r;
    const float v = acc[r] * scale;
    const unsigned short h = f32_to_bf16_rne(v);
    const unsigned short l = f32_to_bf16_rne(v - bf16_bits_to_f32(h));
    dhi[(size_t)crow * C + pcol] = bf16_from_bits(h);
    dlo[(size_t)crow * C + pcol] = bf16_from_bits(l);
  }
}

// ---------------------------------------------------------------------------
// K2: F[n,w,k] = Qn(:,k)^T cov_w Qn(:,k), fused diag extraction.
// LDS holds the Qn slice TRANSPOSED ([col][b], stride 648 bf16 to avoid bank
// conflicts) as planar hi/lo, so B operands are single v16bf LDS loads.
// A operands are single v16bf global loads from the row-major cov planes.
// Each wave loads each A tile once and reuses it for 4 N-tiles (12 WMMAs).
// grid = dim3(16, 75), 256 threads (8 waves).
// ---------------------------------------------------------------------------
constexpr int QT_STRIDE = 648;                              // bf16 per LDS row
constexpr int QT_ELEMS  = 64 * QT_STRIDE;                   // 41472 per plane
constexpr unsigned K2_SMEM = 2 * QT_ELEMS * 2 + 16 * 64 * 4; // 169,984 B

__global__ __launch_bounds__(256) void fmat_kernel(const float* __restrict__ query,
                                                   const bf16_t* __restrict__ cov_hi,
                                                   const bf16_t* __restrict__ cov_lo,
                                                   const float* __restrict__ inv_norm,
                                                   float* __restrict__ Fout) {
  extern __shared__ unsigned char smem_raw[];
  bf16_t* qhi = (bf16_t*)smem_raw;                          // [64][QT_STRIDE]
  bf16_t* qlo = qhi + QT_ELEMS;
  float*  prt = (float*)(smem_raw + (size_t)2 * QT_ELEMS * 2);
  unsigned* qhi_u = (unsigned*)qhi;
  unsigned* qlo_u = (unsigned*)qlo;

  const int tid   = threadIdx.x;
  const int kb    = blockIdx.x;        // 0..15
  const int n     = blockIdx.y;        // 0..74
  const int kbase = kb * 64;

  // Fill transposed planar Qn slice: thread handles column j, rows 2b2,2b2+1.
  const float* Qrow = query + (size_t)n * C * Dd;
  for (int i = tid; i < 64 * (C / 2); i += 256) {
    const int j  = i & 63;
    const int b  = (i >> 6) * 2;
    const float x0 = Qrow[(size_t)b * Dd + kbase + j]       * inv_norm[n * C + b];
    const float x1 = Qrow[(size_t)(b + 1) * Dd + kbase + j] * inv_norm[n * C + b + 1];
    const unsigned short h0 = f32_to_bf16_rne(x0);
    const unsigned short l0 = f32_to_bf16_rne(x0 - bf16_bits_to_f32(h0));
    const unsigned short h1 = f32_to_bf16_rne(x1);
    const unsigned short l1 = f32_to_bf16_rne(x1 - bf16_bits_to_f32(h1));
    const int u = (j * QT_STRIDE + b) >> 1;                 // stride & b even
    qhi_u[u] = (unsigned)h0 | ((unsigned)h1 << 16);
    qlo_u[u] = (unsigned)l0 | ((unsigned)l1 << 16);
  }
  __syncthreads();

  const int wave = tid >> 5;
  const int lane = tid & 31;
  const int half = lane >> 4;
  const int l16  = lane & 15;

  for (int w = 0; w < WAY; ++w) {
    const bf16_t* covWhi = cov_hi + (size_t)w * C * C;
    const bf16_t* covWlo = cov_lo + (size_t)w * C * C;
    float colsum[4] = {0.0f, 0.0f, 0.0f, 0.0f};

    for (int ct = wave; ct < 40; ct += 8) {          // c-tiles, strided by wave
      const int cA = ct * 16 + l16;
      const bf16_t* rowAhi = covWhi + (size_t)cA * C;
      const bf16_t* rowAlo = covWlo + (size_t)cA * C;
      if (ct + 8 < 40) {                             // prefetch next c-tile row
        __builtin_prefetch(rowAhi + (size_t)128 * C, 0, 3);
        __builtin_prefetch(rowAlo + (size_t)128 * C, 0, 3);
      }
      v8f acc[4] = {v8f{}, v8f{}, v8f{}, v8f{}};

      for (int ks = 0; ks < 20; ++ks) {              // K = 640 in chunks of 32
        const int aoff = ks * 32 + half * 16;        // A: elem e -> K=16*half+e
        const v16bf a_hi = *(const v16bf*)(rowAhi + aoff);   // 32B, aligned
        const v16bf a_lo = *(const v16bf*)(rowAlo + aoff);
        const int bb = ks * 32 + half * 16;          // B: elem e -> K=16*half+e
#pragma unroll
        for (int nt = 0; nt < 4; ++nt) {             // reuse A for 4 N-tiles
          const int jcol = nt * 16 + l16;
          const v16bf b_hi = *(const v16bf*)(qhi + jcol * QT_STRIDE + bb);
          const v16bf b_lo = *(const v16bf*)(qlo + jcol * QT_STRIDE + bb);
          acc[nt] = __builtin_amdgcn_wmma_f32_16x16x32_bf16(false, a_hi, false, b_hi, (short)0, acc[nt], false, false);
          acc[nt] = __builtin_amdgcn_wmma_f32_16x16x32_bf16(false, a_hi, false, b_lo, (short)0, acc[nt], false, false);
          acc[nt] = __builtin_amdgcn_wmma_f32_16x16x32_bf16(false, a_lo, false, b_hi, (short)0, acc[nt], false, false);
        }
      }
      // Fused diag: F[k] += sum_c T[c,k] * Qn[c,k]
#pragma unroll
      for (int nt = 0; nt < 4; ++nt) {
        const int jcol = nt * 16 + l16;
#pragma unroll
        for (int r = 0; r < 8; ++r) {                // D: elem r -> M = r + 8*half
          const int crow = ct * 16 + half * 8 + r;
          const float q = (float)qhi[jcol * QT_STRIDE + crow] +
                          (float)qlo[jcol * QT_STRIDE + crow];
          colsum[nt] += acc[nt][r] * q;
        }
      }
    }

    // Deterministic fixed-order reduction of 16 partials per column.
#pragma unroll
    for (int nt = 0; nt < 4; ++nt)
      prt[(wave * 2 + half) * 64 + nt * 16 + l16] = colsum[nt];
    __syncthreads();
    if (tid < 64) {
      float Fv = 0.0f;
#pragma unroll
      for (int q = 0; q < 16; ++q) Fv += prt[q * 64 + tid];
      Fout[((size_t)n * WAY + w) * Dd + kbase + tid] = Fv;
    }
    __syncthreads();
  }
}

// ---------------------------------------------------------------------------
// K3: scores[n,w] = sum_k leakyrelu(F) * conv_w + conv_b.  grid = 375 blocks.
// ---------------------------------------------------------------------------
__global__ __launch_bounds__(256) void score_kernel(const float* __restrict__ F,
                                                    const float* __restrict__ conv_w,
                                                    const float* __restrict__ conv_b,
                                                    float* __restrict__ out) {
  __shared__ float red[256];
  const int nw = blockIdx.x;
  const float* f = F + (size_t)nw * Dd;
  float acc = 0.0f;
  for (int k = threadIdx.x; k < Dd; k += 256) {
    float x = f[k];
    x = (x >= 0.0f) ? x : NEG_SLOPE * x;
    acc += x * conv_w[k];
  }
  red[threadIdx.x] = acc;
  __syncthreads();
  for (int s = 128; s > 0; s >>= 1) {
    if (threadIdx.x < s) red[threadIdx.x] += red[threadIdx.x + s];
    __syncthreads();
  }
  if (threadIdx.x == 0) out[nw] = red[0] + conv_b[0];
}

// ---------------------------------------------------------------------------
extern "C" void kernel_launch(void* const* d_in, const int* in_sizes, int n_in,
                              void* d_out, int out_size, void* d_ws, size_t ws_size,
                              hipStream_t stream) {
  (void)in_sizes; (void)n_in; (void)out_size; (void)ws_size;
  const float* query   = (const float*)d_in[0];
  const float* support = (const float*)d_in[1];
  const float* conv_w  = (const float*)d_in[2];
  const float* conv_b  = (const float*)d_in[3];
  float* out = (float*)d_out;

  // Workspace layout (bytes, all 16B aligned):
  //   mean    @ 0         : WAY*C f32      =    12,800
  //   invnorm @ 12,800    : NQ*C f32       =   192,000
  //   cov_hi  @ 204,800   : WAY*C*C bf16   = 4,096,000
  //   cov_lo  @ 4,300,800 : WAY*C*C bf16   = 4,096,000
  //   F       @ 8,396,800 : NQ*WAY*Dd f32  = 1,536,000   (total ~9.9 MB)
  char* ws = (char*)d_ws;
  float*  mean   = (float*)ws;
  float*  invn   = (float*)(ws + 12800);
  bf16_t* cov_hi = (bf16_t*)(ws + 204800);
  bf16_t* cov_lo = (bf16_t*)(ws + 4300800);
  float*  Fbuf   = (float*)(ws + 8396800);

  // Allow >64KB dynamic LDS (320KB/WGP on gfx1250).
  (void)hipFuncSetAttribute((const void*)fmat_kernel,
                            hipFuncAttributeMaxDynamicSharedMemorySize,
                            (int)K2_SMEM);

  mean_kernel <<<WAY * C,           256, 0, stream>>>(support, mean);
  norm_kernel <<<NQ * C,            256, 0, stream>>>(query, invn);
  cov_kernel  <<<WAY * 40 * 40 / 8, 256, 0, stream>>>(support, mean, cov_hi, cov_lo);
  fmat_kernel <<<dim3(Dd / 64, NQ), 256, K2_SMEM, stream>>>(query, cov_hi, cov_lo, invn, Fbuf);
  score_kernel<<<NQ * WAY,          256, 0, stream>>>(Fbuf, conv_w, conv_b, out);
}
